// model_33225867002038
// MI455X (gfx1250) — matrix operand order repaired and checked by reading, not compiled, not run on hardware
//
#include <hip/hip_runtime.h>
#include <hip/hip_bf16.h>

typedef _Float16 h16;
typedef __attribute__((ext_vector_type(16))) _Float16 v16h;
typedef __attribute__((ext_vector_type(8)))  _Float16 v8h;
typedef __attribute__((ext_vector_type(4)))  _Float16 v4h;
typedef __attribute__((ext_vector_type(8)))  float    v8f;

#define ROWS_PER_BLOCK 128
#define ACTS 136   // actLds row stride in f16 (272B: 16B-aligned, bank-spread)
#define W2S  136   // w2Lds row stride in f16

__device__ __forceinline__ float fast_sigmoid(float x) {
    return __builtin_amdgcn_rcpf(1.0f + __expf(-x));
}
__device__ __forceinline__ float fast_tanh(float x) {
    return fmaf(2.0f, fast_sigmoid(2.0f * x), -1.0f);
}

__device__ __forceinline__ v16h combine16(v8h lo, v8h hi) {
    v16h r;
#pragma unroll
    for (int i = 0; i < 8; ++i) { r[i] = lo[i]; r[i + 8] = hi[i]; }
    return r;
}

__global__ __launch_bounds__(ROWS_PER_BLOCK)
void lstm_mlp_kernel(const float* __restrict__ X,
                     const float* __restrict__ H0,
                     const float* __restrict__ C0,
                     const float* __restrict__ Wih,   // [8,1]
                     const float* __restrict__ Whh,   // [8,2]
                     const float* __restrict__ Bih,   // [8]
                     const float* __restrict__ Bhh,   // [8]
                     const float* __restrict__ W1,    // [128,2]
                     const float* __restrict__ B1,    // [128]
                     const float* __restrict__ W2,    // [5,128]
                     const float* __restrict__ B2,    // [5]
                     float* __restrict__ Out,         // [B,5]
                     int Btotal) {
    // quad tables: slot 128 holds zeros -> hi-half lanes load zeros directly
    __shared__ h16 w1Q[(128 + 8) * 4];   // per neuron: {w0, w1, b1, 0}
    __shared__ h16 aQ[(128 + 8) * 4];    // per row:    {a0, a1, 1.0, 0}
    __shared__ h16 actLds[ROWS_PER_BLOCK * ACTS];
    __shared__ h16 w2Lds[16 * W2S];      // W2 rows (class-major), rows 5..15 = 0

    const int tid = threadIdx.x;
    const long gBase = (long)blockIdx.x * ROWS_PER_BLOCK;

    // ------------- Phase 0: stage FC weights into LDS (f16) ----------------
    {
        v4h q;
        q[0] = (h16)W1[2 * tid];
        q[1] = (h16)W1[2 * tid + 1];
        q[2] = (h16)B1[tid];              // bias folded into K=2 slot
        q[3] = (h16)0;
        *(v4h*)&w1Q[tid * 4] = q;

        if (tid == 0) {
            v4h z;
            z[0] = z[1] = z[2] = z[3] = (h16)0;
            *(v4h*)&w1Q[128 * 4] = z;     // zero quad for hi-half lanes
            *(v4h*)&aQ[128 * 4]  = z;
        }

        // k-column = tid; 16 class rows (5 real, 11 zero); pad cols unused
#pragma unroll
        for (int row = 0; row < 16; ++row) {
            h16 v = (h16)0;
            if (row < 5) v = (h16)W2[row * 128 + tid];
            w2Lds[row * W2S + tid] = v;
        }
    }

    // ------------- Phase 1: per-row LSTM cell (VALU) -----------------------
    {
        long gb = gBase + tid;
        float xv = 0.f, h0a = 0.f, h0b = 0.f, c0a = 0.f, c0b = 0.f;
        if (gb < (long)Btotal) {
            xv  = X[gb];
            h0a = H0[2 * gb];  h0b = H0[2 * gb + 1];
            c0a = C0[2 * gb];  c0b = C0[2 * gb + 1];
        }
        float wih[8], whh[16], bsum[8];
#pragma unroll
        for (int k = 0; k < 8; ++k) { wih[k] = Wih[k]; bsum[k] = Bih[k] + Bhh[k]; }
#pragma unroll
        for (int k = 0; k < 16; ++k) whh[k] = Whh[k];

        float g[8];
#pragma unroll
        for (int k = 0; k < 8; ++k)
            g[k] = fmaf(xv, wih[k], fmaf(h0a, whh[2 * k], fmaf(h0b, whh[2 * k + 1], bsum[k])));

        float i0 = fast_sigmoid(g[0]), i1 = fast_sigmoid(g[1]);
        float f0 = fast_sigmoid(g[2]), f1 = fast_sigmoid(g[3]);
        float gg0 = fast_tanh(g[4]),   gg1 = fast_tanh(g[5]);
        float o0 = fast_sigmoid(g[6]), o1 = fast_sigmoid(g[7]);
        float cn0 = fmaf(f0, c0a, i0 * gg0);
        float cn1 = fmaf(f1, c0b, i1 * gg1);
        float hn0 = o0 * fast_tanh(cn0);
        float hn1 = o1 * fast_tanh(cn1);

        v4h p;
        p[0] = (h16)fmaxf(hn0, 0.0f);
        p[1] = (h16)fmaxf(hn1, 0.0f);
        p[2] = (h16)1.0f;                 // bias multiplier for FC1 K=2 slot
        p[3] = (h16)0;
        *(v4h*)&aQ[tid * 4] = p;
    }

    __syncthreads();

    // ------------- Phase 2: FC1 + FC2 via WMMA (transposed orientation) ----
    const int lane = tid & 31;
    const int wid  = tid >> 5;              // 4 waves
    const int n    = lane & 15;
    const bool hiHalf = (lane & 16) != 0;

    float b2s[5];
#pragma unroll
    for (int j = 0; j < 5; ++j) b2s[j] = B2[j];   // uniform -> scalar loads

    v8h zero8;
#pragma unroll
    for (int i = 0; i < 8; ++i) zero8[i] = (h16)0;

    // hi-half lanes read the zero quad (slot 128) with stride 0
    const int aQidxBase = hiHalf ? 128 : n;       // + rowBase for !hi
    const int w1Base    = hiHalf ? 128 : n;
    const int w1Step    = hiHalf ? 0 : 16;

    for (int tt = 0; tt < 2; ++tt) {
        const int tile = wid * 2 + tt;            // 8 tiles of 16 batch rows
        const int rowBase = tile * 16;
        const int myRow = rowBase + n;

        // ---- FC1: D[neuron, row] = W1 x act^T (+b1 via K=2 trick) --------
        v16h bfrag;
#pragma unroll
        for (int i = 0; i < 16; ++i) bfrag[i] = (h16)0;
        {
            const int qi = hiHalf ? 128 : myRow;  (void)aQidxBase;
            v4h bq = *(const v4h*)&aQ[qi * 4];
#pragma unroll
            for (int i = 0; i < 4; ++i) bfrag[i] = bq[i];
        }

#pragma unroll
        for (int nn = 0; nn < 8; ++nn) {
            v4h wq = *(const v4h*)&w1Q[(w1Base + nn * w1Step) * 4];
            v16h afrag;
#pragma unroll
            for (int i = 0; i < 16; ++i) afrag[i] = (h16)0;
#pragma unroll
            for (int i = 0; i < 4; ++i) afrag[i] = wq[i];

            v8f acc;
#pragma unroll
            for (int i = 0; i < 8; ++i) acc[i] = 0.0f;
            acc = __builtin_amdgcn_wmma_f32_16x16x32_f16(
                false, afrag, false, bfrag, (short)0, acc, false, false);

            // D: lane = row myRow, VGPR r = neuron nn*16 + (hi?8:0) + r.
            // cvt to f16 (pk pairs), ReLU in packed f16, one b128 store.
            v8h packed;
#pragma unroll
            for (int r = 0; r < 8; ++r)
                packed[r] = (h16)acc[r];
            packed = __builtin_elementwise_max(packed, zero8);
            *(v8h*)&actLds[myRow * ACTS + nn * 16 + (hiHalf ? 8 : 0)] = packed;
        }

        // ---- FC2: D[class, row] = W2 x act^T -----------------------------
        v8f acc2;
#pragma unroll
        for (int i = 0; i < 8; ++i) acc2[i] = 0.0f;
        const h16* arow = &w2Lds[n * W2S];          // class row n
        const h16* brow = &actLds[myRow * ACTS];    // activation row
#pragma unroll
        for (int c = 0; c < 4; ++c) {
            const int ka = c * 32 + (hiHalf ? 8 : 0);
            v16h A2 = combine16(*(const v8h*)(arow + ka),
                                *(const v8h*)(arow + ka + 16));
            const int kb = c * 32 + (hiHalf ? 8 : 0);
            v16h B2f = combine16(*(const v8h*)(brow + kb),
                                 *(const v8h*)(brow + kb + 16));
            acc2 = __builtin_amdgcn_wmma_f32_16x16x32_f16(
                false, A2, false, B2f, (short)0, acc2, false, false);
        }

        // D: lane = batch row, VGPR r = class; classes 8..15 (hi lanes) are
        // zero padding -> only lanes 0-15 store the 5 real classes.
        long grow = gBase + myRow;
        if (!hiHalf && grow < (long)Btotal) {
            float* op = Out + grow * 5;
#pragma unroll
            for (int j = 0; j < 5; ++j)
                op[j] = acc2[j] + b2s[j];
        }
    }
}

extern "C" void kernel_launch(void* const* d_in, const int* in_sizes, int n_in,
                              void* d_out, int out_size, void* d_ws, size_t ws_size,
                              hipStream_t stream) {
    const float* X   = (const float*)d_in[0];
    const float* H0  = (const float*)d_in[1];
    const float* C0  = (const float*)d_in[2];
    const float* Wih = (const float*)d_in[3];
    const float* Whh = (const float*)d_in[4];
    const float* Bih = (const float*)d_in[5];
    const float* Bhh = (const float*)d_in[6];
    const float* W1  = (const float*)d_in[7];
    const float* B1  = (const float*)d_in[8];
    const float* W2  = (const float*)d_in[9];
    const float* B2  = (const float*)d_in[10];
    float* Out = (float*)d_out;

    const int Btotal = in_sizes[0];                 // INPUT_SIZE == 1
    const int grid = (Btotal + ROWS_PER_BLOCK - 1) / ROWS_PER_BLOCK;
    lstm_mlp_kernel<<<grid, ROWS_PER_BLOCK, 0, stream>>>(
        X, H0, C0, Wih, Whh, Bih, Bhh, W1, B1, W2, B2, Out, Btotal);
}
